// ManualAttention_2250562863803
// MI455X (gfx1250) — matrix operand order repaired and checked by reading, not compiled, not run on hardware
//
#include <hip/hip_runtime.h>

// ---- types for CDNA5 WMMA ----
typedef __attribute__((ext_vector_type(16))) __bf16        v16bf;
typedef __attribute__((ext_vector_type(8)))  float         v8f;
typedef __attribute__((ext_vector_type(4)))  float         v4f;
typedef __attribute__((ext_vector_type(8)))  unsigned int  v8u;
typedef __attribute__((ext_vector_type(4)))  unsigned int  v4u;

__device__ __forceinline__ unsigned short f2bf(float f) {
    unsigned int u = __float_as_uint(f);
    u += 0x7FFFu + ((u >> 16) & 1u);   // round-to-nearest-even
    return (unsigned short)(u >> 16);
}
__device__ __forceinline__ unsigned int pk2(float lo, float hi) {
    return (unsigned int)f2bf(lo) | ((unsigned int)f2bf(hi) << 16);
}

__device__ __forceinline__ v8f wmma_bf16(v16bf a, v16bf b, v8f c) {
    return __builtin_amdgcn_wmma_f32_16x16x32_bf16(false, a, false, b,
                                                   (short)0, c, false, false);
}

// cross-half exchange (lane i <-> lane i+16), pure VALU (v_permlanex16_b32)
__device__ __forceinline__ float permx16(float x) {
    int xi = __builtin_bit_cast(int, x);
    int r  = __builtin_amdgcn_permlanex16(xi, xi, 0x76543210, 0xFEDCBA98, false, false);
    return __builtin_bit_cast(float, r);
}

// B-operand: 16 contiguous bf16 (32B, 16B-aligned) from LDS
__device__ __forceinline__ v16bf ldB(const unsigned short* p) {
    v4u x = *(const v4u*)p;
    v4u y = *(const v4u*)(p + 8);
    v8u u;
    u[0] = x[0]; u[1] = x[1]; u[2] = x[2]; u[3] = x[3];
    u[4] = y[0]; u[5] = y[1]; u[6] = y[2]; u[7] = y[3];
    return __builtin_bit_cast(v16bf, u);
}
// A-operand: per-lane two 16B row chunks at +16*half and +16*half+8 (ushort units)
__device__ __forceinline__ v16bf ldA(const unsigned short* row, int half) {
    v4u x = *(const v4u*)(row + 16 * half);
    v4u y = *(const v4u*)(row + 16 * half + 8);
    v8u u;
    u[0] = x[0]; u[1] = x[1]; u[2] = x[2]; u[3] = x[3];
    u[4] = y[0]; u[5] = y[1]; u[6] = y[2]; u[7] = y[3];
    return __builtin_bit_cast(v16bf, u);
}

// Transposed flash attention: S^T = K*Q^T, O^T = V^T*P^T, double-buffered staging
// with loads issued after the score tiles die (short register live ranges).
// grid = (S/128, B*H), block = 128 (4 waves); each wave owns 32 queries x 64 dims.
__global__ __launch_bounds__(128)
void fa_wmma_kernel(const float* __restrict__ q,
                    const float* __restrict__ k,
                    const float* __restrict__ v,
                    float* __restrict__ out)
{
    constexpr int   S      = 2048;
    constexpr int   D      = 64;
    constexpr int   NT     = S / 32;
    constexpr float QSCALE = 0.125f * 1.44269504088896340736f; // 1/sqrt(64)*log2(e)
    constexpr int   KP     = 72;  // K tile LDS row stride (ushorts)
    constexpr int   VP     = 40;  // V^T LDS row stride
    constexpr int   PP     = 40;  // P^T scratch row stride

    __shared__ __align__(16) unsigned short lds_k [2][32 * KP];       // double-buffered K tiles
    __shared__ __align__(16) unsigned short lds_vt[2][64 * VP];       // double-buffered V^T tiles
    __shared__ __align__(16) unsigned short lds_p [4][2][16 * PP];    // per-wave, per-qt P^T scratch

    const int tid    = threadIdx.x;
    const int wave   = tid >> 5;
    const int lane   = tid & 31;
    const int half   = lane >> 4;
    const int lanelo = lane & 15;

    const int    bh    = blockIdx.y;
    const size_t base  = (size_t)bh * S * D;
    const int    qrow0 = blockIdx.x * 128 + wave * 32;

    // ---------- Q rows straight into B-layout (lane n = query, contiguous dim chunk) ----------
    v16bf bq[2][2];
    #pragma unroll
    for (int qt = 0; qt < 2; ++qt) {
        const float4* qr4 = (const float4*)(q + base + (size_t)(qrow0 + qt * 16 + lanelo) * D);
        #pragma unroll
        for (int ks = 0; ks < 2; ++ks) {
            const int f0 = (ks * 32 + 16 * half) >> 2;
            float4 a0 = qr4[f0], a1 = qr4[f0 + 1], a2 = qr4[f0 + 2], a3 = qr4[f0 + 3];
            v8u u;
            u[0] = pk2(a0.x * QSCALE, a0.y * QSCALE);
            u[1] = pk2(a0.z * QSCALE, a0.w * QSCALE);
            u[2] = pk2(a1.x * QSCALE, a1.y * QSCALE);
            u[3] = pk2(a1.z * QSCALE, a1.w * QSCALE);
            u[4] = pk2(a2.x * QSCALE, a2.y * QSCALE);
            u[5] = pk2(a2.z * QSCALE, a2.w * QSCALE);
            u[6] = pk2(a3.x * QSCALE, a3.y * QSCALE);
            u[7] = pk2(a3.z * QSCALE, a3.w * QSCALE);
            bq[qt][ks] = __builtin_bit_cast(v16bf, u);
        }
    }

    // ---------- flash-attention state ----------
    v8f   o[2][4];
    float m_[2] = { -1.0e30f, -1.0e30f };
    float l_[2] = { 0.0f, 0.0f };
    #pragma unroll
    for (int qt = 0; qt < 2; ++qt)
        #pragma unroll
        for (int dt = 0; dt < 4; ++dt)
            #pragma unroll
            for (int j = 0; j < 8; ++j) o[qt][dt][j] = 0.0f;

    // staging roles: thread -> (key row pair, 8-dim segment)
    const int r2  = tid & 15;
    const int seg = (tid >> 4) * 8;
    const float* kbase = k + base + (size_t)(2*r2) * D + seg;
    const float* vbase = v + base + (size_t)(2*r2) * D + seg;

    const v8f zc = {};

    // ---------- pipeline prologue: stage tile 0 (buffer 0) ----------
    {
        const float4* kp0 = (const float4*)kbase;
        const float4* kp1 = (const float4*)(kbase + D);
        const float4* vp0 = (const float4*)vbase;
        const float4* vp1 = (const float4*)(vbase + D);
        float4 ka0 = kp0[0], ka1 = kp0[1], kb0 = kp1[0], kb1 = kp1[1];
        float4 va0 = vp0[0], va1 = vp0[1], vb0 = vp1[0], vb1 = vp1[1];
        v4u kv0, kv1;
        kv0[0] = pk2(ka0.x, ka0.y); kv0[1] = pk2(ka0.z, ka0.w);
        kv0[2] = pk2(ka1.x, ka1.y); kv0[3] = pk2(ka1.z, ka1.w);
        kv1[0] = pk2(kb0.x, kb0.y); kv1[1] = pk2(kb0.z, kb0.w);
        kv1[2] = pk2(kb1.x, kb1.y); kv1[3] = pk2(kb1.z, kb1.w);
        *(v4u*)&lds_k[0][(2*r2)     * KP + seg] = kv0;
        *(v4u*)&lds_k[0][(2*r2 + 1) * KP + seg] = kv1;
        unsigned short* vt = &lds_vt[0][0];
        *(unsigned int*)&vt[(seg + 0) * VP + 2*r2] = pk2(va0.x, vb0.x);
        *(unsigned int*)&vt[(seg + 1) * VP + 2*r2] = pk2(va0.y, vb0.y);
        *(unsigned int*)&vt[(seg + 2) * VP + 2*r2] = pk2(va0.z, vb0.z);
        *(unsigned int*)&vt[(seg + 3) * VP + 2*r2] = pk2(va0.w, vb0.w);
        *(unsigned int*)&vt[(seg + 4) * VP + 2*r2] = pk2(va1.x, vb1.x);
        *(unsigned int*)&vt[(seg + 5) * VP + 2*r2] = pk2(va1.y, vb1.y);
        *(unsigned int*)&vt[(seg + 6) * VP + 2*r2] = pk2(va1.z, vb1.z);
        *(unsigned int*)&vt[(seg + 7) * VP + 2*r2] = pk2(va1.w, vb1.w);
    }
    __syncthreads();

    #pragma unroll 1
    for (int kt = 0; kt < NT; ++kt) {
        const int  cur  = kt & 1;
        const bool more = (kt + 1 < NT);

        // ---------- S^T = K * Q^T  (M = key, N = query) ----------
        v8f c[2][2];
        #pragma unroll
        for (int mt = 0; mt < 2; ++mt) {
            const unsigned short* krow = &lds_k[cur][(mt * 16 + lanelo) * KP];
            v16bf ak0 = ldA(krow, half);           // dims 0..31
            c[mt][0] = wmma_bf16(ak0, bq[0][0], zc);
            c[mt][1] = wmma_bf16(ak0, bq[1][0], zc);
            v16bf ak1 = ldA(krow + 32, half);      // dims 32..63
            c[mt][0] = wmma_bf16(ak1, bq[0][1], c[mt][0]);
            c[mt][1] = wmma_bf16(ak1, bq[1][1], c[mt][1]);
        }

        // ---------- lane-local online softmax; vectorized P^T spill ----------
        #pragma unroll
        for (int qt = 0; qt < 2; ++qt) {
            float mx = fmaxf(c[0][qt][0], c[0][qt][1]);
            #pragma unroll
            for (int j = 2; j < 8; ++j) mx = fmaxf(mx, c[0][qt][j]);
            #pragma unroll
            for (int j = 0; j < 8; ++j) mx = fmaxf(mx, c[1][qt][j]);
            mx = fmaxf(mx, permx16(mx));

            const float mn = fmaxf(m_[qt], mx);
            const float al = __builtin_exp2f(m_[qt] - mn);
            m_[qt] = mn;

            float rs = 0.0f;
            #pragma unroll
            for (int mt = 0; mt < 2; ++mt) {
                v4u pkv;
                #pragma unroll
                for (int i = 0; i < 4; ++i) {
                    const float p0 = __builtin_exp2f(c[mt][qt][2*i]     - mn);
                    const float p1 = __builtin_exp2f(c[mt][qt][2*i + 1] - mn);
                    rs += p0 + p1;
                    pkv[i] = pk2(p0, p1);
                }
                *(v4u*)&lds_p[wave][qt][lanelo * PP + mt * 16 + 8 * half] = pkv;
            }
            rs += permx16(rs);
            l_[qt] = l_[qt] * al + rs;

            #pragma unroll
            for (int dt = 0; dt < 4; ++dt)
                #pragma unroll
                for (int j = 0; j < 8; ++j) o[qt][dt][j] *= al;
        }

        // wave-level LDS RAW fence before cross-lane re-read of P^T
        asm volatile("s_wait_dscnt 0" ::: "memory");

        // ---------- issue next tile's global loads (score tiles are dead now) ----------
        float4 ka0, ka1, kb0, kb1, va0, va1, vb0, vb1;
        if (more) {
            const size_t off = (size_t)(kt + 1) * 32 * D;
            const float4* kp0 = (const float4*)(kbase + off);
            const float4* kp1 = (const float4*)(kbase + off + D);
            const float4* vp0 = (const float4*)(vbase + off);
            const float4* vp1 = (const float4*)(vbase + off + D);
            ka0 = kp0[0]; ka1 = kp0[1]; kb0 = kp1[0]; kb1 = kp1[1];
            va0 = vp0[0]; va1 = vp0[1]; vb0 = vp1[0]; vb1 = vp1[1];
        }

        // ---------- P^T in B-layout (lane n = query, contiguous key chunk) ----------
        v16bf pb[2];
        pb[0] = ldB(&lds_p[wave][0][lanelo * PP + 16 * half]);
        pb[1] = ldB(&lds_p[wave][1][lanelo * PP + 16 * half]);

        // ---------- O^T += V^T * P^T (shared V^T A-operands across sub-tiles) ----------
        #pragma unroll
        for (int dt = 0; dt < 4; ++dt) {
            v16bf av = ldA(&lds_vt[cur][(dt * 16 + lanelo) * VP], half);
            o[0][dt] = wmma_bf16(av, pb[0], o[0][dt]);
            o[1][dt] = wmma_bf16(av, pb[1], o[1][dt]);
        }

        // ---------- stage next tile into the other buffer, single barrier ----------
        if (more) {
            v4u kv0, kv1;
            kv0[0] = pk2(ka0.x, ka0.y); kv0[1] = pk2(ka0.z, ka0.w);
            kv0[2] = pk2(ka1.x, ka1.y); kv0[3] = pk2(ka1.z, ka1.w);
            kv1[0] = pk2(kb0.x, kb0.y); kv1[1] = pk2(kb0.z, kb0.w);
            kv1[2] = pk2(kb1.x, kb1.y); kv1[3] = pk2(kb1.z, kb1.w);
            *(v4u*)&lds_k[cur ^ 1][(2*r2)     * KP + seg] = kv0;
            *(v4u*)&lds_k[cur ^ 1][(2*r2 + 1) * KP + seg] = kv1;
            unsigned short* vt = &lds_vt[cur ^ 1][0];
            *(unsigned int*)&vt[(seg + 0) * VP + 2*r2] = pk2(va0.x, vb0.x);
            *(unsigned int*)&vt[(seg + 1) * VP + 2*r2] = pk2(va0.y, vb0.y);
            *(unsigned int*)&vt[(seg + 2) * VP + 2*r2] = pk2(va0.z, vb0.z);
            *(unsigned int*)&vt[(seg + 3) * VP + 2*r2] = pk2(va0.w, vb0.w);
            *(unsigned int*)&vt[(seg + 4) * VP + 2*r2] = pk2(va1.x, vb1.x);
            *(unsigned int*)&vt[(seg + 5) * VP + 2*r2] = pk2(va1.y, vb1.y);
            *(unsigned int*)&vt[(seg + 6) * VP + 2*r2] = pk2(va1.z, vb1.z);
            *(unsigned int*)&vt[(seg + 7) * VP + 2*r2] = pk2(va1.w, vb1.w);
        }
        __syncthreads();
    }

    // ---------- normalize (v_rcp_f32) and store: 16B vector stores, row-contiguous ----------
    #pragma unroll
    for (int qt = 0; qt < 2; ++qt) {
        const float inv = __builtin_amdgcn_rcpf(l_[qt]);
        float* op = out + base + (size_t)(qrow0 + qt * 16 + lanelo) * D;
        #pragma unroll
        for (int dt = 0; dt < 4; ++dt) {
            v8f s = o[qt][dt];
            #pragma unroll
            for (int j = 0; j < 8; ++j) s[j] *= inv;
            v4f lo = __builtin_shufflevector(s, s, 0, 1, 2, 3);
            v4f hi = __builtin_shufflevector(s, s, 4, 5, 6, 7);
            *(v4f*)&op[dt * 16 + 8 * half]     = lo;
            *(v4f*)&op[dt * 16 + 8 * half + 4] = hi;
        }
    }
}

extern "C" void kernel_launch(void* const* d_in, const int* in_sizes, int n_in,
                              void* d_out, int out_size, void* d_ws, size_t ws_size,
                              hipStream_t stream) {
    (void)in_sizes; (void)n_in; (void)out_size; (void)d_ws; (void)ws_size;
    const float* q = (const float*)d_in[0];
    const float* k = (const float*)d_in[1];
    const float* v = (const float*)d_in[2];
    float* out = (float*)d_out;

    constexpr int B = 4, H = 16, S = 2048;
    dim3 grid(S / 128, B * H);  // 16 query blocks x 64 (b,h) pairs
    dim3 block(128);            // 4 waves, each owning 32 query rows
    fa_wmma_kernel<<<grid, block, 0, stream>>>(q, k, v, out);
}
